// SlotAttention_60352880443937
// MI455X (gfx1250) — matrix operand order repaired and checked by reading, not compiled, not run on hardware
//
#include <hip/hip_runtime.h>

// ---------------------------------------------------------------------------
// CDNA5 (gfx1250) SlotAttention: bf16 WMMA GEMMs (all NT), async-LDS double
// buffering, LN / slot-softmax / transpose / elementwise helpers.
// ---------------------------------------------------------------------------

typedef __attribute__((ext_vector_type(16))) __bf16 v16bf;
typedef __attribute__((ext_vector_type(8)))  float  v8f;

#define BKS 64                   // k-step per LDS tile
#define LDS_STRIDE 72            // 64 bf16 payload + 8 pad (144B rows, 16B aligned)

#if __has_builtin(__builtin_amdgcn_global_load_async_to_lds_b128)
#define HAVE_ASYNC_LDS 1
#else
#define HAVE_ASYNC_LDS 0
#endif

#if HAVE_ASYNC_LDS
typedef int v4i_t __attribute__((vector_size(16)));
typedef __attribute__((address_space(1))) v4i_t* as1_v4i;
typedef __attribute__((address_space(3))) v4i_t* as3_v4i;

__device__ __forceinline__ void async_cp16(const __bf16* g, __bf16* l) {
    __builtin_amdgcn_global_load_async_to_lds_b128(
        (as1_v4i)(void*)const_cast<__bf16*>(g),
        (as3_v4i)(void*)l, 0, 0);
}
template <int Nleft>
__device__ __forceinline__ void wait_async() {
#if __has_builtin(__builtin_amdgcn_s_wait_asynccnt)
    __builtin_amdgcn_s_wait_asynccnt(Nleft);
#else
    asm volatile("s_wait_asynccnt %0" :: "i"(Nleft));
#endif
}
#endif

// One LDS tile pair -> 16 WMMA (2 k-chunks of 32, 4x2 subtiles per wave).
// Wave tile is always 64 (M) x 32 (N); mBase/nBase place it in the block tile.
__device__ __forceinline__ void compute_tile(const __bf16* __restrict__ As,
                                             const __bf16* __restrict__ Bs,
                                             int mBase, int nBase, int g, int ln16,
                                             v8f (&acc)[4][2])
{
    #pragma unroll
    for (int kc = 0; kc < 2; ++kc) {
        const int kb = kc * 32;
        v16bf af[4];
        #pragma unroll
        for (int mi = 0; mi < 4; ++mi) {
            int m = mBase + mi * 16 + ln16;
            unsigned int* ap = (unsigned int*)&af[mi];
            #pragma unroll
            for (int v = 0; v < 8; ++v) {
                int kk = kb + (v < 4 ? 2 * v : 2 * v + 8) + 8 * g;   // pairs (kk,kk+1)
                ap[v] = *(const unsigned int*)(&As[m * LDS_STRIDE + kk]);
            }
        }
        v16bf bfr[2];
        #pragma unroll
        for (int ni = 0; ni < 2; ++ni) {
            int n = nBase + ni * 16 + ln16;
            unsigned int* bp = (unsigned int*)&bfr[ni];
            #pragma unroll
            for (int v = 0; v < 8; ++v) {
                int kk = kb + (v < 4 ? 2 * v : 2 * v + 8) + 8 * g;
                bp[v] = *(const unsigned int*)(&Bs[n * LDS_STRIDE + kk]);
            }
        }
        #pragma unroll
        for (int mi = 0; mi < 4; ++mi)
            #pragma unroll
            for (int ni = 0; ni < 2; ++ni)
                acc[mi][ni] = __builtin_amdgcn_wmma_f32_16x16x32_bf16(
                    false, af[mi], false, bfr[ni], (short)0, acc[mi][ni], false, false);
    }
}

// ---------------------------------------------------------------------------
// NT GEMM: C[M,N] = act(scale * A[M,Kd] @ Bt^T + bias), Bt is [N,Kd] row-major.
// Block tile BM x BN (BM*BN == 16384), 8 waves of 64x32 wave tiles:
//   <128,128>: waves 2x4 (general M)
//   < 64,256>: waves 1x8 (M == 64 attention GEMMs, no wasted rows)
// Batched via blockIdx.z. Ragged M handled by row clamp + guarded stores.
// ---------------------------------------------------------------------------
template <int BM, int BN, bool HAS_BIAS, bool GELU, bool OUT_BF16>
__global__ __launch_bounds__(256)
void gemm_wmma_nt(const __bf16* __restrict__ A, const __bf16* __restrict__ Bt,
                  const float* __restrict__ bias, void* __restrict__ Cout,
                  int M, int N, int Kd, int lda, int ldb, int ldc,
                  long long sA, long long sB, long long sC, float scale)
{
    constexpr int WAVES_M = BM / 64;          // 2 or 1
    constexpr int WAVES_N = 8 / WAVES_M;      // 4 or 8
    static_assert(BM * BN == 16384 && BN == WAVES_N * 32, "tile config");
    constexpr int NA = BM / 32;               // uint4 staging loads per thread (A)
    constexpr int NB = BN / 32;               // uint4 staging loads per thread (B)

    __shared__ __align__(16) __bf16 As[2][BM * LDS_STRIDE];
    __shared__ __align__(16) __bf16 Bs[2][BN * LDS_STRIDE];

    const int z = blockIdx.z;
    A  += (long long)z * sA;
    Bt += (long long)z * sB;
    char* Cbase = (char*)Cout;

    const int tid   = threadIdx.x;
    const int wave  = tid >> 5;
    const int lane  = tid & 31;
    const int waveM = wave % WAVES_M;
    const int waveN = wave / WAVES_M;
    const int g     = lane >> 4;
    const int ln16  = lane & 15;
    const int mBase = waveM * 64;
    const int nBase = waveN * 32;

    const int bM = blockIdx.y * BM;
    const int bN = blockIdx.x * BN;

    // staging coords: rows x 64 bf16, 8 uint4 per row
    const int srow = tid >> 3;            // 0..31, +32 per chunk
    const int scol = (tid & 7) << 3;      // 0,8,...,56

    v8f acc[4][2] = {};
    const int nTiles = Kd / BKS;

#if HAVE_ASYNC_LDS
    // ---------------- async-to-LDS double-buffered pipeline ----------------
    {   // issue tile 0 -> buf 0
        #pragma unroll
        for (int i = 0; i < NA; ++i) {
            int row = srow + i * 32;
            int ar  = bM + row; if (ar >= M) ar = M - 1;
            async_cp16(A + (long long)ar * lda + scol, &As[0][row * LDS_STRIDE + scol]);
        }
        #pragma unroll
        for (int i = 0; i < NB; ++i) {
            int row = srow + i * 32;
            async_cp16(Bt + (long long)(bN + row) * ldb + scol, &Bs[0][row * LDS_STRIDE + scol]);
        }
    }
    int cur = 0;
    for (int t = 0; t < nTiles; ++t) {
        const int nxt = cur ^ 1;
        if (t + 1 < nTiles) {
            const int k0 = (t + 1) * BKS;
            #pragma unroll
            for (int i = 0; i < NA; ++i) {
                int row = srow + i * 32;
                int ar  = bM + row; if (ar >= M) ar = M - 1;
                async_cp16(A + (long long)ar * lda + k0 + scol, &As[nxt][row * LDS_STRIDE + scol]);
            }
            #pragma unroll
            for (int i = 0; i < NB; ++i) {
                int row = srow + i * 32;
                async_cp16(Bt + (long long)(bN + row) * ldb + k0 + scol, &Bs[nxt][row * LDS_STRIDE + scol]);
            }
            wait_async<NA + NB>();   // tile t's DMAs done; tile t+1 still in flight
        } else {
            wait_async<0>();
        }
        __syncthreads();             // all waves' tile-t data landed in LDS
        compute_tile(As[cur], Bs[cur], mBase, nBase, g, ln16, acc);
        __syncthreads();             // done reading buf cur; safe to overwrite next iter
        cur = nxt;
    }
#else
    // ---------------- fallback: register-staged single buffer ----------------
    for (int t = 0; t < nTiles; ++t) {
        const int k0 = t * BKS;
        uint4 ra[NA], rb[NB];
        #pragma unroll
        for (int i = 0; i < NA; ++i) {
            int row = srow + i * 32;
            int ar  = bM + row; if (ar >= M) ar = M - 1;
            ra[i] = *(const uint4*)(A + (long long)ar * lda + k0 + scol);
        }
        #pragma unroll
        for (int i = 0; i < NB; ++i) {
            int row = srow + i * 32;
            rb[i] = *(const uint4*)(Bt + (long long)(bN + row) * ldb + k0 + scol);
        }
        __syncthreads();
        #pragma unroll
        for (int i = 0; i < NA; ++i)
            *(uint4*)(&As[0][(srow + i * 32) * LDS_STRIDE + scol]) = ra[i];
        #pragma unroll
        for (int i = 0; i < NB; ++i)
            *(uint4*)(&Bs[0][(srow + i * 32) * LDS_STRIDE + scol]) = rb[i];
        __syncthreads();
        compute_tile(As[0], Bs[0], mBase, nBase, g, ln16, acc);
    }
    __syncthreads();
#endif

    // ---- epilogue: scale, bias, gelu, store f32/bf16 ----
    #pragma unroll
    for (int mi = 0; mi < 4; ++mi) {
        #pragma unroll
        for (int ni = 0; ni < 2; ++ni) {
            int n = bN + nBase + ni * 16 + ln16;
            #pragma unroll
            for (int r = 0; r < 8; ++r) {
                int m = bM + mBase + mi * 16 + 8 * g + r;
                if (m < M) {
                    float val = acc[mi][ni][r] * scale;
                    if (HAS_BIAS) val += bias[n];
                    if (GELU)     val = 0.5f * val * (1.0f + erff(val * 0.70710678118654752f));
                    long long cidx = (long long)z * sC + (long long)m * ldc + n;
                    if (OUT_BF16) ((__bf16*)Cbase)[cidx] = (__bf16)val;
                    else          ((float*)Cbase)[cidx]  = val;
                }
            }
        }
    }
}

// ---------------------------------------------------------------------------
// Transposes (tiled through LDS, coalesced both sides)
// ---------------------------------------------------------------------------
__global__ __launch_bounds__(256)
void transpose_f32_to_bf16_kernel(const float* __restrict__ in, __bf16* __restrict__ out,
                                  int R, int C)   // in [R][C] -> out [C][R]
{
    __shared__ float tile[32][33];
    const int bx = blockIdx.x * 32;           // col base
    const int by = blockIdx.y * 32;           // row base
    const int tx = threadIdx.x & 31;
    const int ty = threadIdx.x >> 5;          // 0..7
    #pragma unroll
    for (int j = 0; j < 4; ++j)
        tile[ty + j * 8][tx] = in[(long long)(by + ty + j * 8) * C + bx + tx];
    __syncthreads();
    #pragma unroll
    for (int j = 0; j < 4; ++j)
        out[(long long)(bx + ty + j * 8) * R + by + tx] = (__bf16)tile[tx][ty + j * 8];
}

__global__ __launch_bounds__(256)
void transpose_bf16_kernel(const __bf16* __restrict__ in, __bf16* __restrict__ out,
                           int R, int C)   // per batch z: in [R][C] -> out [C][R]
{
    __shared__ __bf16 tile[32][34];
    const long long boff = (long long)blockIdx.z * R * C;
    in += boff; out += boff;
    const int bx = blockIdx.x * 32;
    const int by = blockIdx.y * 32;
    const int tx = threadIdx.x & 31;
    const int ty = threadIdx.x >> 5;
    #pragma unroll
    for (int j = 0; j < 4; ++j)
        tile[ty + j * 8][tx] = in[(long long)(by + ty + j * 8) * C + bx + tx];
    __syncthreads();
    #pragma unroll
    for (int j = 0; j < 4; ++j)
        out[(long long)(bx + ty + j * 8) * R + by + tx] = tile[tx][ty + j * 8];
}

// ---------------------------------------------------------------------------
// Elementwise helpers
// ---------------------------------------------------------------------------
__global__ void f32_to_bf16_kernel(const float* __restrict__ in, __bf16* __restrict__ out, long long n) {
    long long i = (long long)blockIdx.x * blockDim.x + threadIdx.x;
    long long stride = (long long)gridDim.x * blockDim.x;
    for (; i < n; i += stride) out[i] = (__bf16)in[i];
}

__global__ void broadcast_slots_kernel(const float* __restrict__ seed, float* __restrict__ out, long long n) {
    long long i = (long long)blockIdx.x * blockDim.x + threadIdx.x;
    long long stride = (long long)gridDim.x * blockDim.x;
    for (; i < n; i += stride) out[i] = seed[i & 65535];   // 64*1024 = 65536
}

__global__ void add_kernel(const float* __restrict__ a, const float* __restrict__ b,
                           float* __restrict__ dst, long long n) {
    long long i = (long long)blockIdx.x * blockDim.x + threadIdx.x;
    long long stride = (long long)gridDim.x * blockDim.x;
    for (; i < n; i += stride) dst[i] = a[i] + b[i];
}

// ---------------------------------------------------------------------------
// Row LayerNorm: in f32 [rows, D] -> out bf16 [rows, D]
// ---------------------------------------------------------------------------
__global__ __launch_bounds__(256)
void layernorm_kernel(const float* __restrict__ in, const float* __restrict__ gam,
                      const float* __restrict__ bet, __bf16* __restrict__ out, int D)
{
    __shared__ float red[256];
    const long long row = blockIdx.x;
    const float* x = in + row * D;
    const int tid = threadIdx.x;

    float s = 0.0f;
    for (int i = tid; i < D; i += 256) s += x[i];
    red[tid] = s; __syncthreads();
    for (int off = 128; off > 0; off >>= 1) {
        if (tid < off) red[tid] += red[tid + off];
        __syncthreads();
    }
    float mu = red[0] / (float)D;
    __syncthreads();

    float s2 = 0.0f;
    for (int i = tid; i < D; i += 256) { float d = x[i] - mu; s2 += d * d; }
    red[tid] = s2; __syncthreads();
    for (int off = 128; off > 0; off >>= 1) {
        if (tid < off) red[tid] += red[tid + off];
        __syncthreads();
    }
    float rstd = rsqrtf(red[0] / (float)D + 1e-5f);

    __bf16* o = out + row * D;
    for (int i = tid; i < D; i += 256)
        o[i] = (__bf16)((x[i] - mu) * rstd * gam[i] + bet[i]);
}

// ---------------------------------------------------------------------------
// Softmax over the slot axis: logits f32 [B][64][N] -> attn bf16 [B][64][N]
// ---------------------------------------------------------------------------
__global__ __launch_bounds__(256)
void softmax_slots_kernel(const float* __restrict__ logits, __bf16* __restrict__ attn, int N)
{
    const int b = blockIdx.y;
    const int n = blockIdx.x * 256 + threadIdx.x;
    const float* L = logits + (long long)b * 64 * N + n;
    __bf16* O = attn + (long long)b * 64 * N + n;

    float c[64];
    float mx = -3.402823466e+38f;
    #pragma unroll
    for (int k = 0; k < 64; ++k) {
        c[k] = L[(long long)k * N];
        mx = fmaxf(mx, c[k]);
    }
    float sum = 0.0f;
    #pragma unroll
    for (int k = 0; k < 64; ++k) { c[k] = __expf(c[k] - mx); sum += c[k]; }
    float inv = 1.0f / sum;
    #pragma unroll
    for (int k = 0; k < 64; ++k) O[(long long)k * N] = (__bf16)(c[k] * inv);
}

// ---------------------------------------------------------------------------
// Host orchestration
// ---------------------------------------------------------------------------
static inline size_t ws_alloc(size_t& off, size_t bytes) {
    size_t cur = off;
    off += (bytes + 255) & ~(size_t)255;
    return cur;
}

extern "C" void kernel_launch(void* const* d_in, const int* in_sizes, int n_in,
                              void* d_out, int out_size, void* d_ws, size_t ws_size,
                              hipStream_t stream)
{
    (void)in_sizes; (void)n_in; (void)out_size; (void)ws_size;

    constexpr int B = 8, N = 4096, DIN = 1024, K = 64, D = 1024;
    constexpr long long BN = (long long)B * N;          // 32768 token rows
    constexpr long long BK = (long long)B * K;          // 512 slot rows

    const float* inputs     = (const float*)d_in[0];
    const float* W_in       = (const float*)d_in[1];
    const float* b_in       = (const float*)d_in[2];
    const float* slots_seed = (const float*)d_in[3];
    const float* ln_in_g    = (const float*)d_in[4];
    const float* ln_in_b    = (const float*)d_in[5];
    const float* ln_sl_g    = (const float*)d_in[6];
    const float* ln_sl_b    = (const float*)d_in[7];
    const float* ln_mlp_g   = (const float*)d_in[8];
    const float* ln_mlp_b   = (const float*)d_in[9];
    const float* Wq         = (const float*)d_in[10];
    const float* Wk         = (const float*)d_in[11];
    const float* Wv         = (const float*)d_in[12];
    const float* W1         = (const float*)d_in[13];
    const float* b1         = (const float*)d_in[14];
    const float* W2         = (const float*)d_in[15];
    const float* b2         = (const float*)d_in[16];
    float* out = (float*)d_out;

    char* ws = (char*)d_ws;
    size_t off = 0;
    __bf16* in_bf  = (__bf16*)(ws + ws_alloc(off, BN * DIN * 2));
    __bf16* winT   = (__bf16*)(ws + ws_alloc(off, (size_t)DIN * D * 2));      // [D][DIN]
    __bf16* wqT    = (__bf16*)(ws + ws_alloc(off, (size_t)D * D * 2));        // [D][D]
    __bf16* wkT    = (__bf16*)(ws + ws_alloc(off, (size_t)D * D * 2));
    __bf16* wvT    = (__bf16*)(ws + ws_alloc(off, (size_t)D * D * 2));
    __bf16* w1T    = (__bf16*)(ws + ws_alloc(off, (size_t)D * 2 * D * 2));    // [2D][D]
    __bf16* w2T    = (__bf16*)(ws + ws_alloc(off, (size_t)2 * D * D * 2));    // [D][2D]
    float*  xtmp   = (float*)(ws + ws_alloc(off, BN * D * 4));
    __bf16* x_bf   = (__bf16*)(ws + ws_alloc(off, BN * D * 2));
    __bf16* k_bf   = (__bf16*)(ws + ws_alloc(off, BN * D * 2));
    __bf16* v_bf   = (__bf16*)(ws + ws_alloc(off, BN * D * 2));
    __bf16* vT     = (__bf16*)(ws + ws_alloc(off, BN * D * 2));               // [B][D][N]
    float*  slots  = (float*)(ws + ws_alloc(off, BK * D * 4));
    __bf16* sl_bf  = (__bf16*)(ws + ws_alloc(off, BK * D * 2));
    __bf16* q_bf   = (__bf16*)(ws + ws_alloc(off, BK * D * 2));
    float*  logits = (float*)(ws + ws_alloc(off, (size_t)B * K * N * 4));
    __bf16* attn   = (__bf16*)(ws + ws_alloc(off, (size_t)B * K * N * 2));
    float*  upd    = (float*)(ws + ws_alloc(off, BK * D * 4));
    __bf16* h_bf   = (__bf16*)(ws + ws_alloc(off, BK * D * 2));
    __bf16* m1_bf  = (__bf16*)(ws + ws_alloc(off, BK * 2 * D * 2));
    float*  m2     = (float*)(ws + ws_alloc(off, BK * D * 4));

    const dim3 blk(256);
    auto ew_grid = [](long long n) { long long g = (n + 255) / 256; if (g > 4096) g = 4096; return dim3((unsigned)g); };

    // ---- 1) convert inputs; transpose-convert all weights ----
    f32_to_bf16_kernel<<<ew_grid(BN * DIN), blk, 0, stream>>>(inputs, in_bf, BN * DIN);
    transpose_f32_to_bf16_kernel<<<dim3(D / 32, DIN / 32), blk, 0, stream>>>(W_in, winT, DIN, D);
    transpose_f32_to_bf16_kernel<<<dim3(D / 32, D / 32), blk, 0, stream>>>(Wq, wqT, D, D);
    transpose_f32_to_bf16_kernel<<<dim3(D / 32, D / 32), blk, 0, stream>>>(Wk, wkT, D, D);
    transpose_f32_to_bf16_kernel<<<dim3(D / 32, D / 32), blk, 0, stream>>>(Wv, wvT, D, D);
    transpose_f32_to_bf16_kernel<<<dim3(2 * D / 32, D / 32), blk, 0, stream>>>(W1, w1T, D, 2 * D);
    transpose_f32_to_bf16_kernel<<<dim3(D / 32, 2 * D / 32), blk, 0, stream>>>(W2, w2T, 2 * D, D);

    // ---- 2) xtmp = inputs @ W_in + b_in ; x = LN(xtmp) ----
    {
        dim3 grid(D / 128, (unsigned)(BN / 128), 1);
        gemm_wmma_nt<128, 128, true, false, false><<<grid, blk, 0, stream>>>(
            in_bf, winT, b_in, xtmp, (int)BN, D, DIN, DIN, DIN, D, 0, 0, 0, 1.0f);
    }
    layernorm_kernel<<<dim3((unsigned)BN), blk, 0, stream>>>(xtmp, ln_in_g, ln_in_b, x_bf, D);

    // ---- 3) k = x @ Wk ; v = x @ Wv ; vT = transpose(v) ----
    {
        dim3 grid(D / 128, (unsigned)(BN / 128), 1);
        gemm_wmma_nt<128, 128, false, false, true><<<grid, blk, 0, stream>>>(
            x_bf, wkT, nullptr, k_bf, (int)BN, D, D, D, D, D, 0, 0, 0, 1.0f);
        gemm_wmma_nt<128, 128, false, false, true><<<grid, blk, 0, stream>>>(
            x_bf, wvT, nullptr, v_bf, (int)BN, D, D, D, D, D, 0, 0, 0, 1.0f);
    }
    transpose_bf16_kernel<<<dim3(D / 32, N / 32, B), blk, 0, stream>>>(v_bf, vT, N, D);

    // ---- 4) slots = broadcast(slots_seed) ----
    broadcast_slots_kernel<<<ew_grid(BK * D), blk, 0, stream>>>(slots_seed, slots, BK * D);

    const float scale = 0.03125f;  // D^-0.5 = 1/32

    for (int it = 0; it < 3; ++it) {
        // slots_ln -> q
        layernorm_kernel<<<dim3((unsigned)BK), blk, 0, stream>>>(slots, ln_sl_g, ln_sl_b, sl_bf, D);
        {
            dim3 grid(D / 128, (unsigned)(BK / 128), 1);
            gemm_wmma_nt<128, 128, false, false, true><<<grid, blk, 0, stream>>>(
                sl_bf, wqT, nullptr, q_bf, (int)BK, D, D, D, D, D, 0, 0, 0, 1.0f);
        }
        // logits[b] = q[b] @ k[b]^T * scale   (Bt = k, already [N][Kd]); M=64 tile
        {
            dim3 grid(N / 256, 1, B);
            gemm_wmma_nt<64, 256, false, false, false><<<grid, blk, 0, stream>>>(
                q_bf, k_bf, nullptr, logits, K, N, D, D, D, N,
                (long long)K * D, (long long)N * D, (long long)K * N, scale);
        }
        // attn = softmax over slots
        softmax_slots_kernel<<<dim3(N / 256, B), blk, 0, stream>>>(logits, attn, N);
        // updates[b] = attn[b] @ v[b]   (Bt = vT[b] = [D][N]); M=64 tile
        {
            dim3 grid(D / 256, 1, B);
            gemm_wmma_nt<64, 256, false, false, false><<<grid, blk, 0, stream>>>(
                attn, vT, nullptr, upd, K, D, N, N, N, D,
                (long long)K * N, (long long)D * N, (long long)K * D, 1.0f);
        }
        // h = LN(updates) ; mlp = gelu(h@W1+b1)@W2+b2 ; slots += mlp
        layernorm_kernel<<<dim3((unsigned)BK), blk, 0, stream>>>(upd, ln_mlp_g, ln_mlp_b, h_bf, D);
        {
            dim3 grid(2 * D / 128, (unsigned)(BK / 128), 1);
            gemm_wmma_nt<128, 128, true, true, true><<<grid, blk, 0, stream>>>(
                h_bf, w1T, b1, m1_bf, (int)BK, 2 * D, D, D, D, 2 * D, 0, 0, 0, 1.0f);
        }
        {
            dim3 grid(D / 128, (unsigned)(BK / 128), 1);
            gemm_wmma_nt<128, 128, true, false, false><<<grid, blk, 0, stream>>>(
                m1_bf, w2T, b2, m2, (int)BK, D, 2 * D, 2 * D, 2 * D, D, 0, 0, 0, 1.0f);
        }
        float* dst = (it == 2) ? out : slots;
        add_kernel<<<ew_grid(BK * D), blk, 0, stream>>>(slots, m2, dst, BK * D);
    }
}
